// MultipleKeyAttention_88648124989648
// MI455X (gfx1250) — matrix operand order repaired and checked by reading, not compiled, not run on hardware
//
#include <hip/hip_runtime.h>
#include <hip/hip_bf16.h>

// Problem constants
#define kN  2
#define kT  1024
#define kS  1024
#define kB  4
#define kE  1024
#define kH  16
#define kHD 64

// HD^-0.5 * log2(e): softmax done in base 2 (single v_exp_f32 per prob)
#define kQScale 0.18033688011112042f

typedef __attribute__((ext_vector_type(16))) __bf16 v16bf;
typedef __attribute__((ext_vector_type(8)))  float  v8f;

__device__ __forceinline__ __bf16 f2bf(float f) { return (__bf16)f; }

__device__ __forceinline__ v8f wmma_bf(v16bf a, v16bf b, v8f c) {
  // D(16x16,f32) = A(16x32,bf16) x B(32x16,bf16) + C
  return __builtin_amdgcn_wmma_f32_16x16x32_bf16(false, a, false, b, (short)0, c,
                                                 false, false);
}

// A-fragment (16x32 MxK) from bf16 row-major matrix (row stride ld).
// lane li(0..15)=row M; hi=0: K {0..7,16..23}; hi=1: K {8..15,24..31}
__device__ __forceinline__ v16bf load_a_bf(const __bf16* __restrict__ X, int ld,
                                           int m0, int k0, int lane) {
  int li = lane & 15, hi = lane >> 4;
  const __bf16* r = X + (size_t)(m0 + li) * ld + (k0 + 8 * hi);
  v16bf a;
#pragma unroll
  for (int e = 0; e < 8; ++e) a[e] = r[e];
#pragma unroll
  for (int e = 0; e < 8; ++e) a[8 + e] = r[16 + e];
  return a;
}

// B-fragment (32x16 KxN) where B[k][n] = M[n][k], M bf16 row-major stride ld.
// lane li = column N; hi=0: K {0..7,16..23}; hi=1: K {8..15,24..31}
__device__ __forceinline__ v16bf load_b_bfT(const __bf16* __restrict__ M, int ld,
                                            int n0, int k0, int lane) {
  int li = lane & 15, hi = lane >> 4;
  const __bf16* r = M + (size_t)(n0 + li) * ld + (k0 + 8 * hi);
  v16bf b;
#pragma unroll
  for (int e = 0; e < 8; ++e) b[e] = r[e];
#pragma unroll
  for (int e = 0; e < 8; ++e) b[8 + e] = r[16 + e];
  return b;
}

// ---------------- f32 -> bf16 bulk conversion (8 elems/thread)
__global__ void cvt_kernel(const float* __restrict__ src, __bf16* __restrict__ dst) {
  size_t i = ((size_t)blockIdx.x * blockDim.x + threadIdx.x) * 8;
#pragma unroll
  for (int e = 0; e < 8; ++e) dst[i + e] = f2bf(src[i + e]);
}

// ---------------- Q projection: q[n,b,h,t,hd] = (query @ Wq[n].T + bq[n]) * kQScale
// 32 rows x 64 cols per wave
__global__ void qproj_kernel(const __bf16* __restrict__ query,
                             const __bf16* __restrict__ Wq,
                             const float* __restrict__ bq,
                             __bf16* __restrict__ qout) {
  const int MT = (kT * kB) / 32, FT = kE / 64;   // 128, 16
  int lane = threadIdx.x & 31, wave = threadIdx.x >> 5;
  int task = blockIdx.x * (blockDim.x >> 5) + wave;
  int n = task / (MT * FT);
  int rem = task % (MT * FT);
  int m0 = (rem / FT) * 32, f0 = (rem % FT) * 64;
  const __bf16* Wn = Wq + (size_t)n * kE * kE;

  v8f acc[2][4] = {};
  for (int k0 = 0; k0 < kE; k0 += 32) {
    v16bf a0 = load_a_bf(query, kE, m0, k0, lane);
    v16bf a1 = load_a_bf(query, kE, m0 + 16, k0, lane);
#pragma unroll
    for (int j = 0; j < 4; ++j) {
      v16bf bfr = load_b_bfT(Wn, kE, f0 + 16 * j, k0, lane);
      acc[0][j] = wmma_bf(a0, bfr, acc[0][j]);
      acc[1][j] = wmma_bf(a1, bfr, acc[1][j]);
    }
  }
  int li = lane & 15, hi = lane >> 4;
  int h = f0 >> 6;
#pragma unroll
  for (int j = 0; j < 4; ++j) {
    int f = f0 + 16 * j + li;
    float bias = bq[n * kE + f];
    int hd = f & 63;
#pragma unroll
    for (int r = 0; r < 2; ++r)
#pragma unroll
      for (int e = 0; e < 8; ++e) {
        int m = m0 + 16 * r + 8 * hi + e;
        int t = m >> 2, b = m & 3;
        float y = (acc[r][j][e] + bias) * kQScale;
        qout[((((size_t)n * kB + b) * kH + h) * kT + t) * kHD + hd] = f2bf(y);
      }
  }
}

// ---------------- K and V projections (shared A fragments), 32x64 per wave
// k[n,b,h,s,hd]; v stored transposed: vT[n,b,h,hd,s]
__global__ void kvproj_kernel(const __bf16* __restrict__ keys,
                              const __bf16* __restrict__ Wk,
                              const float* __restrict__ bk,
                              const __bf16* __restrict__ Wv,
                              const float* __restrict__ bv,
                              __bf16* __restrict__ kout,
                              __bf16* __restrict__ vTout) {
  const int MT = (kS * kB) / 32, FT = kE / 64;   // 128, 16
  int lane = threadIdx.x & 31, wave = threadIdx.x >> 5;
  int task = blockIdx.x * (blockDim.x >> 5) + wave;
  int n = task / (MT * FT);
  int rem = task % (MT * FT);
  int m0 = (rem / FT) * 32, f0 = (rem % FT) * 64;
  const __bf16* Xn  = keys + (size_t)n * kS * kB * kE;
  const __bf16* Wkn = Wk + (size_t)n * kE * kE;
  const __bf16* Wvn = Wv + (size_t)n * kE * kE;

  v8f acck[2][4] = {}, accv[2][4] = {};
  for (int k0 = 0; k0 < kE; k0 += 32) {
    v16bf a0 = load_a_bf(Xn, kE, m0, k0, lane);
    v16bf a1 = load_a_bf(Xn, kE, m0 + 16, k0, lane);
#pragma unroll
    for (int j = 0; j < 4; ++j) {
      v16bf bk_f = load_b_bfT(Wkn, kE, f0 + 16 * j, k0, lane);
      acck[0][j] = wmma_bf(a0, bk_f, acck[0][j]);
      acck[1][j] = wmma_bf(a1, bk_f, acck[1][j]);
      v16bf bv_f = load_b_bfT(Wvn, kE, f0 + 16 * j, k0, lane);
      accv[0][j] = wmma_bf(a0, bv_f, accv[0][j]);
      accv[1][j] = wmma_bf(a1, bv_f, accv[1][j]);
    }
  }
  int li = lane & 15, hi = lane >> 4;
  int h = f0 >> 6;
#pragma unroll
  for (int j = 0; j < 4; ++j) {
    int f = f0 + 16 * j + li;
    float biask = bk[n * kE + f];
    float biasv = bv[n * kE + f];
    int hd = f & 63;
#pragma unroll
    for (int r = 0; r < 2; ++r)
#pragma unroll
      for (int e = 0; e < 8; ++e) {
        int m = m0 + 16 * r + 8 * hi + e;
        int s = m >> 2, b = m & 3;
        size_t base = (((size_t)n * kB + b) * kH + h);
        kout[(base * kS + s) * kHD + hd]  = f2bf(acck[r][j][e] + biask);
        vTout[(base * kHD + hd) * kS + s] = f2bf(accv[r][j][e] + biasv);
      }
  }
}

// ---------------- Flash attention over concatenated sources, 64-key S-tiles
#define ATT_WAVES 4
__global__ void attn_kernel(const __bf16* __restrict__ qbf,
                            const __bf16* __restrict__ kbf,
                            const __bf16* __restrict__ vT,
                            const unsigned char* __restrict__ masks,
                            __bf16* __restrict__ attn_out) {
  __shared__ __bf16 pLds[ATT_WAVES][16][64];
  int lane = threadIdx.x & 31, wave = threadIdx.x >> 5;
  int task = blockIdx.x * ATT_WAVES + wave;
  const int TT = kT / 16;
  int tt = task % TT;
  int bh = task / TT;
  int h = bh % kH, b = bh / kH;
  int t0 = tt * 16;
  int li = lane & 15, hi = lane >> 4;
  const float NEG_INF = -__builtin_inff();

  v8f oacc[4] = {};
  float mrun[8], lrun[8];
#pragma unroll
  for (int j = 0; j < 8; ++j) { mrun[j] = NEG_INF; lrun[j] = 0.f; }

  for (int n = 0; n < kN; ++n) {
    size_t base = (((size_t)n * kB + b) * kH + h);
    const __bf16* qb = qbf + base * kT * kHD;
    const __bf16* kb = kbf + base * kS * kHD;
    const __bf16* vb = vT + base * kHD * kS;
    const unsigned char* mk = masks + ((size_t)n * kB + b) * kS;
    v16bf qa0 = load_a_bf(qb, kHD, t0, 0, lane);
    v16bf qa1 = load_a_bf(qb, kHD, t0, 32, lane);

    for (int s0 = 0; s0 < kS; s0 += 64) {
      // scores for 64 keys: four 16-col tiles, K-loop over HD=64
      v8f sc[4];
      bool msk[4];
#pragma unroll
      for (int u = 0; u < 4; ++u) {
        sc[u] = v8f{};
        sc[u] = wmma_bf(qa0, load_b_bfT(kb, kHD, s0 + 16 * u, 0, lane), sc[u]);
        sc[u] = wmma_bf(qa1, load_b_bfT(kb, kHD, s0 + 16 * u, 32, lane), sc[u]);
        msk[u] = mk[s0 + 16 * u + li] != 0;   // per-lane key column
      }
#pragma unroll
      for (int u = 0; u < 4; ++u)
        if (msk[u])
#pragma unroll
          for (int j = 0; j < 8; ++j) sc[u][j] = NEG_INF;
      // row max across the 16-lane half-group (rows t = 8*hi + j)
      float tmax[8];
#pragma unroll
      for (int j = 0; j < 8; ++j)
        tmax[j] = fmaxf(fmaxf(sc[0][j], sc[1][j]), fmaxf(sc[2][j], sc[3][j]));
#pragma unroll
      for (int off = 1; off < 16; off <<= 1)
#pragma unroll
        for (int j = 0; j < 8; ++j)
          tmax[j] = fmaxf(tmax[j], __shfl_xor(tmax[j], off, 32));
      // online softmax update (base-2 domain)
      float corr[8], psum[8];
      float p[4][8];
#pragma unroll
      for (int j = 0; j < 8; ++j) {
        float nm = fmaxf(mrun[j], tmax[j]);
        corr[j] = (mrun[j] == nm) ? 1.f : __builtin_amdgcn_exp2f(mrun[j] - nm);
        mrun[j] = nm;
        float ps = 0.f;
#pragma unroll
        for (int u = 0; u < 4; ++u) {
          p[u][j] = (nm == NEG_INF) ? 0.f : __builtin_amdgcn_exp2f(sc[u][j] - nm);
          ps += p[u][j];
        }
        psum[j] = ps;
      }
#pragma unroll
      for (int off = 1; off < 16; off <<= 1)
#pragma unroll
        for (int j = 0; j < 8; ++j)
          psum[j] += __shfl_xor(psum[j], off, 32);
#pragma unroll
      for (int j = 0; j < 8; ++j) lrun[j] = lrun[j] * corr[j] + psum[j];
#pragma unroll
      for (int dt = 0; dt < 4; ++dt)
#pragma unroll
        for (int j = 0; j < 8; ++j) oacc[dt][j] *= corr[j];
      // transpose P (D-layout, col-per-lane) -> A-fragments via LDS
      __syncthreads();
#pragma unroll
      for (int u = 0; u < 4; ++u)
#pragma unroll
        for (int j = 0; j < 8; ++j)
          pLds[wave][8 * hi + j][16 * u + li] = f2bf(p[u][j]);
      __syncthreads();
      v16bf pa0, pa1;
#pragma unroll
      for (int e = 0; e < 8; ++e) {
        pa0[e]     = pLds[wave][li][8 * hi + e];
        pa0[8 + e] = pLds[wave][li][16 + 8 * hi + e];
        pa1[e]     = pLds[wave][li][32 + 8 * hi + e];
        pa1[8 + e] = pLds[wave][li][48 + 8 * hi + e];
      }
      // O += P(16x64) x V(64x16) per 16-wide d tile
#pragma unroll
      for (int dt = 0; dt < 4; ++dt) {
        oacc[dt] = wmma_bf(pa0, load_b_bfT(vb, kS, dt * 16, s0, lane), oacc[dt]);
        oacc[dt] = wmma_bf(pa1, load_b_bfT(vb, kS, dt * 16, s0 + 32, lane), oacc[dt]);
      }
    }
  }
  // finalize and store bf16 [t, b, h*64 + d]
  float inv[8];
#pragma unroll
  for (int j = 0; j < 8; ++j) inv[j] = (lrun[j] > 0.f) ? 1.f / lrun[j] : 0.f;
#pragma unroll
  for (int dt = 0; dt < 4; ++dt)
#pragma unroll
    for (int j = 0; j < 8; ++j) {
      int t = t0 + 8 * hi + j;
      attn_out[((size_t)t * kB + b) * kE + h * 64 + dt * 16 + li] =
          f2bf(oacc[dt][j] * inv[j]);
    }
}

// ---------------- Output projection: out = attn @ Wo.T + bo (f32 out), 32x64/wave
__global__ void oproj_kernel(const __bf16* __restrict__ attn,
                             const __bf16* __restrict__ Wo,
                             const float* __restrict__ bo,
                             float* __restrict__ out) {
  const int FT = kE / 64;
  int lane = threadIdx.x & 31, wave = threadIdx.x >> 5;
  int task = blockIdx.x * (blockDim.x >> 5) + wave;
  int m0 = (task / FT) * 32, f0 = (task % FT) * 64;

  v8f acc[2][4] = {};
  for (int k0 = 0; k0 < kE; k0 += 32) {
    v16bf a0 = load_a_bf(attn, kE, m0, k0, lane);
    v16bf a1 = load_a_bf(attn, kE, m0 + 16, k0, lane);
#pragma unroll
    for (int j = 0; j < 4; ++j) {
      v16bf bfr = load_b_bfT(Wo, kE, f0 + 16 * j, k0, lane);
      acc[0][j] = wmma_bf(a0, bfr, acc[0][j]);
      acc[1][j] = wmma_bf(a1, bfr, acc[1][j]);
    }
  }
  int li = lane & 15, hi = lane >> 4;
#pragma unroll
  for (int j = 0; j < 4; ++j) {
    float bias = bo[f0 + 16 * j + li];
#pragma unroll
    for (int r = 0; r < 2; ++r)
#pragma unroll
      for (int e = 0; e < 8; ++e) {
        int m = m0 + 16 * r + 8 * hi + e;
        out[(size_t)m * kE + f0 + 16 * j + li] = acc[r][j][e] + bias;
      }
  }
}

extern "C" void kernel_launch(void* const* d_in, const int* in_sizes, int n_in,
                              void* d_out, int out_size, void* d_ws, size_t ws_size,
                              hipStream_t stream) {
  const float* query = (const float*)d_in[0];
  const float* keys  = (const float*)d_in[1];
  const unsigned char* masks = (const unsigned char*)d_in[2];
  const float* Wq = (const float*)d_in[3];
  const float* bq = (const float*)d_in[4];
  const float* Wk = (const float*)d_in[5];
  const float* bk = (const float*)d_in[6];
  const float* Wv = (const float*)d_in[7];
  const float* bv = (const float*)d_in[8];
  const float* Wo = (const float*)d_in[9];
  const float* bo = (const float*)d_in[10];
  float* out = (float*)d_out;

  char* ws = (char*)d_ws;
  size_t off = 0;
  const size_t MB = 1024 * 1024;
  __bf16* qbf     = (__bf16*)(ws + off); off += 16 * MB;  // [n,b,h,t,hd]
  __bf16* kbf     = (__bf16*)(ws + off); off += 16 * MB;  // [n,b,h,s,hd]
  __bf16* vTb     = (__bf16*)(ws + off); off += 16 * MB;  // [n,b,h,hd,s]
  __bf16* attn    = (__bf16*)(ws + off); off += 8 * MB;   // [t,b,e]
  __bf16* qry_bf  = (__bf16*)(ws + off); off += 8 * MB;   // [t,b,e]
  __bf16* keys_bf = (__bf16*)(ws + off); off += 16 * MB;  // [n,s,b,e]
  __bf16* Wq_bf   = (__bf16*)(ws + off); off += 4 * MB;
  __bf16* Wk_bf   = (__bf16*)(ws + off); off += 4 * MB;
  __bf16* Wv_bf   = (__bf16*)(ws + off); off += 4 * MB;
  __bf16* Wo_bf   = (__bf16*)(ws + off); off += 2 * MB;

  // bf16 conversion pre-pass (8 elems per thread, 256 threads per block)
  cvt_kernel<<<(kT * kB * kE) / (8 * 256), 256, 0, stream>>>(query, qry_bf);
  cvt_kernel<<<(kN * kS * kB * kE) / (8 * 256), 256, 0, stream>>>(keys, keys_bf);
  cvt_kernel<<<(kN * kE * kE) / (8 * 256), 256, 0, stream>>>(Wq, Wq_bf);
  cvt_kernel<<<(kN * kE * kE) / (8 * 256), 256, 0, stream>>>(Wk, Wk_bf);
  cvt_kernel<<<(kN * kE * kE) / (8 * 256), 256, 0, stream>>>(Wv, Wv_bf);
  cvt_kernel<<<(kE * kE) / (8 * 256), 256, 0, stream>>>(Wo, Wo_bf);

  // Q projection: 2*128*16 = 4096 wave-tiles, 4 waves/block
  qproj_kernel<<<1024, 128, 0, stream>>>(qry_bf, Wq_bf, bq, qbf);
  // K+V projection: 4096 wave-tiles
  kvproj_kernel<<<1024, 128, 0, stream>>>(keys_bf, Wk_bf, bk, Wv_bf, bv, kbf, vTb);
  // Flash attention: B*H*(T/16) = 4096 wave-tiles, 4 waves/block
  attn_kernel<<<1024, 128, 0, stream>>>(qbf, kbf, vTb, masks, attn);
  // Output projection: 128*16 = 2048 wave-tiles
  oproj_kernel<<<512, 128, 0, stream>>>(attn, Wo_bf, bo, out);
}
